// CTEBlock_32298154066020
// MI455X (gfx1250) — matrix operand order repaired and checked by reading, not compiled, not run on hardware
//
#include <hip/hip_runtime.h>
#include <hip/hip_bf16.h>
#include <math.h>

// ---------------- types / WMMA helpers ----------------
typedef _Float16 v8h  __attribute__((ext_vector_type(8)));
typedef _Float16 v16h __attribute__((ext_vector_type(16)));
typedef float    v8f  __attribute__((ext_vector_type(8)));
typedef float    v4f  __attribute__((ext_vector_type(4)));

#define WMMA_F16(a, b, c) \
  __builtin_amdgcn_wmma_f32_16x16x32_f16(false, (a), false, (b), (short)0, (c), false, false)

// problem constants
#define BATCH   32
#define NTOK    320
#define LENS_T  64
#define LENS_S  256
#define DIM     768
#define NHEADS  12
#define HEAD_D  64
#define MLP_H   3072
#define NKEEP   180          // ceil(0.7*256)
#define NREM    76
#define NNEW    244          // 64 + 180
#define MTOK    (BATCH*NTOK) // 10240
#define MNEW    (BATCH*NNEW) // 7808

// LDS layout for attention (dynamic shared memory)
#define VT_LD    336                       // f16 row stride for V^T (672 B, 32B aligned)
#define S_LD     324                       // f32 row stride for score strip (16B aligned, breaks bank pattern)
#define VT_BYTES (HEAD_D * VT_LD * 2)      // 43008
#define S_BYTES  (16 * S_LD * 4)           // 20736 per wave
#define INV_BYTES (8 * 16 * 4)             // 512
#define ATT_SMEM (VT_BYTES + 8 * S_BYTES + INV_BYTES + NTOK)  // + mask bytes

// A-operand tile loader: 16x32 f16, row-major source with leading dim ld.
// lane m = lane&15 (row), kh = lane>>4. K index of element j = (j&7)+8*kh+16*(j>>3).
__device__ __forceinline__ v16h ldA(const _Float16* __restrict__ base, int ld, int k0, int m, int kh) {
  const _Float16* p = base + (size_t)m * ld + k0 + kh * 8;
  v8h lo = *(const v8h*)p;
  v8h hi = *(const v8h*)(p + 16);
  v16h r;
#pragma unroll
  for (int i = 0; i < 8; ++i) { r[i] = lo[i]; r[i + 8] = hi[i]; }
  return r;
}

// B-operand tile loader: 32x16 (KxN) where memory holds Bt[n][k] row-major (weight layout).
// lane n = lane&15 (col), elements j -> k = k0 + (j&7) + 8*kh + 16*(j>>3) : two 16B chunks.
__device__ __forceinline__ v16h ldB(const _Float16* __restrict__ baseT, int ld, int k0, int n, int kh) {
  const _Float16* p = baseT + (size_t)n * ld + k0 + kh * 8;
  v8h lo = *(const v8h*)p;
  v8h hi = *(const v8h*)(p + 16);
  v16h r;
#pragma unroll
  for (int i = 0; i < 8; ++i) { r[i] = lo[i]; r[i + 8] = hi[i]; }
  return r;
}

// ---------------- elementwise f32 -> f16 ----------------
__global__ void f32_to_f16_kernel(const float* __restrict__ in, _Float16* __restrict__ out, int n) {
  int i = blockIdx.x * 256 + threadIdx.x;
  if (i < n) out[i] = (_Float16)in[i];
}

// ---------------- LayerNorm (row = 768) -> f16 ----------------
__global__ __launch_bounds__(256) void layernorm_f16_kernel(
    const float* __restrict__ x, const float* __restrict__ g, const float* __restrict__ bt,
    _Float16* __restrict__ y) {
  int row = blockIdx.x;
  int tid = threadIdx.x;
  const float* xr = x + (size_t)row * DIM;
  float v0 = xr[tid], v1 = xr[tid + 256], v2 = xr[tid + 512];
  float s = v0 + v1 + v2;
  float q = v0 * v0 + v1 * v1 + v2 * v2;
#pragma unroll
  for (int off = 16; off; off >>= 1) {
    s += __shfl_xor(s, off, 32);
    q += __shfl_xor(q, off, 32);
  }
  __shared__ float ss[8], sq[8];
  int wv = tid >> 5;
  if ((tid & 31) == 0) { ss[wv] = s; sq[wv] = q; }
  __syncthreads();
  float ts = 0.f, tq = 0.f;
#pragma unroll
  for (int w = 0; w < 8; ++w) { ts += ss[w]; tq += sq[w]; }
  float mean = ts * (1.0f / DIM);
  float var  = tq * (1.0f / DIM) - mean * mean;
  float inv  = rsqrtf(var + 1e-5f);
  _Float16* yr = y + (size_t)row * DIM;
  yr[tid]       = (_Float16)((v0 - mean) * inv * g[tid]       + bt[tid]);
  yr[tid + 256] = (_Float16)((v1 - mean) * inv * g[tid + 256] + bt[tid + 256]);
  yr[tid + 512] = (_Float16)((v2 - mean) * inv * g[tid + 512] + bt[tid + 512]);
}

// ---------------- generic WMMA GEMM: out = act(A @ W^T + bias) (+ residual) ----------------
// A: [M,K] f16 row-major.  W: [N,K] f16 row-major (B operand as-is).
// grid = (M/64, N/256), block = 256 (8 waves in 2x4), each wave computes a 32x64 tile
// (2 M-tiles x 4 N-tiles -> 8 WMMAs per K-step from 6 tile loads).
template <bool OUT16, bool GELU>
__global__ __launch_bounds__(256) void gemm_wmma_kernel(
    const _Float16* __restrict__ A, const _Float16* __restrict__ W,
    const float* __restrict__ bias, const float* __restrict__ residual,
    _Float16* __restrict__ out16, float* __restrict__ out32,
    int M, int N, int K) {
  int lane = threadIdx.x & 31;
  int wave = threadIdx.x >> 5;
  int m16  = lane & 15;
  int kh   = lane >> 4;
  int row0 = blockIdx.x * 64 + (wave >> 2) * 32;
  int col0 = blockIdx.y * 256 + (wave & 3) * 64;

  v8f c[2][4] = {};
  const _Float16* Abase = A + (size_t)row0 * K;
  for (int k0 = 0; k0 < K; k0 += 32) {
    // prefetch next K-chunk (global_prefetch_b8)
    if (k0 + 32 < K) {
      __builtin_prefetch(Abase + (size_t)m16 * K + k0 + 32, 0, 1);
      __builtin_prefetch(W + (size_t)(col0 + m16) * K + k0 + 32, 0, 1);
      __builtin_prefetch(W + (size_t)(col0 + 32 + m16) * K + k0 + 32, 0, 1);
    }
    v16h a0 = ldA(Abase,                  K, k0, m16, kh);
    v16h a1 = ldA(Abase + (size_t)16 * K, K, k0, m16, kh);
    v16h b0 = ldB(W + (size_t)(col0)      * K, K, k0, m16, kh);
    v16h b1 = ldB(W + (size_t)(col0 + 16) * K, K, k0, m16, kh);
    v16h b2 = ldB(W + (size_t)(col0 + 32) * K, K, k0, m16, kh);
    v16h b3 = ldB(W + (size_t)(col0 + 48) * K, K, k0, m16, kh);
    c[0][0] = WMMA_F16(a0, b0, c[0][0]);
    c[0][1] = WMMA_F16(a0, b1, c[0][1]);
    c[0][2] = WMMA_F16(a0, b2, c[0][2]);
    c[0][3] = WMMA_F16(a0, b3, c[0][3]);
    c[1][0] = WMMA_F16(a1, b0, c[1][0]);
    c[1][1] = WMMA_F16(a1, b1, c[1][1]);
    c[1][2] = WMMA_F16(a1, b2, c[1][2]);
    c[1][3] = WMMA_F16(a1, b3, c[1][3]);
  }

#pragma unroll
  for (int i = 0; i < 2; ++i) {
#pragma unroll
    for (int j = 0; j < 4; ++j) {
      int col = col0 + j * 16 + m16;
      float bv = bias ? bias[col] : 0.0f;
#pragma unroll
      for (int r = 0; r < 8; ++r) {
        int row = row0 + i * 16 + r + 8 * kh;
        float v = c[i][j][r] + bv;
        if (GELU) v = 0.5f * v * (1.0f + erff(v * 0.70710678118654752f));
        size_t idx = (size_t)row * N + col;
        if (residual) v += residual[idx];
        if (OUT16) out16[idx] = (_Float16)v;
        else       out32[idx] = v;
      }
    }
  }
}

// ---------------- fused attention per (batch, head) ----------------
// qkv16: [B*320, 2304] f16 (q | k | v, each 12 heads x 64)
// corrmap: [B,12,320,320] f32 (output attn)   ao16: [B*320, 768] f16 attention output
// Dynamic LDS: V^T (shared) + per-wave 16x320 f32 score strip + row-inv broadcast + mask bytes.
__global__ __launch_bounds__(256) void attention_kernel(
    const _Float16* __restrict__ qkv, const unsigned char* __restrict__ mask,
    float* __restrict__ corrmap, _Float16* __restrict__ ao16) {
  extern __shared__ char smem_raw[];
  _Float16* vT      = (_Float16*)smem_raw;                                 // [64][VT_LD]
  float* Sall       = (float*)(smem_raw + VT_BYTES);                       // 8 x [16][S_LD]
  float* invAll     = (float*)(smem_raw + VT_BYTES + 8 * S_BYTES);         // 8 x [16]
  unsigned char* mL = (unsigned char*)(smem_raw + VT_BYTES + 8 * S_BYTES + INV_BYTES); // [320]

  int b = blockIdx.x / NHEADS;
  int h = blockIdx.x % NHEADS;
  const size_t rowBase = (size_t)b * NTOK;

  // cooperatively stage V transposed + mask bytes into LDS
  for (int i = threadIdx.x; i < NTOK * HEAD_D; i += 256) {
    int t = i >> 6, d = i & 63;
    vT[d * VT_LD + t] = qkv[(rowBase + t) * 2304 + 1536 + h * HEAD_D + d];
  }
  for (int i = threadIdx.x; i < NTOK; i += 256) mL[i] = mask[b * NTOK + i];
  __syncthreads();

  int lane = threadIdx.x & 31;
  int wave = threadIdx.x >> 5;
  int n16  = lane & 15;
  int kh   = lane >> 4;
  float* S      = Sall + wave * (16 * S_LD);
  float* rowinv = invAll + wave * 16;

  const _Float16* kb = qkv + rowBase * 2304 + 768 + h * HEAD_D;

  for (int strip = wave; strip < NTOK / 16; strip += 8) {
    int row0 = strip * 16;

    // Q tiles (16x64 => two K=32 A-tiles)
    const _Float16* qb = qkv + (rowBase + row0) * 2304 + h * HEAD_D;
    v16h q0 = ldA(qb, 2304, 0, n16, kh);
    v16h q1 = ldA(qb, 2304, 32, n16, kh);

    // ---- pass 1: S = scale*(Q K^T) with mask, store to LDS, track row max ----
    float rmax[8];
#pragma unroll
    for (int r = 0; r < 8; ++r) rmax[r] = -3.0e38f;
#pragma unroll 2
    for (int nc = 0; nc < 20; ++nc) {
      v16h bk0 = ldB(kb, 2304, 0, nc * 16 + n16, kh);
      v16h bk1 = ldB(kb, 2304, 32, nc * 16 + n16, kh);
      v8f acc = {};
      acc = WMMA_F16(q0, bk0, acc);
      acc = WMMA_F16(q1, bk1, acc);
      int key = nc * 16 + n16;
      bool mk = mL[key] != 0;
#pragma unroll
      for (int r = 0; r < 8; ++r) {
        float v = mk ? -1.0e9f : acc[r] * 0.125f;
        S[(r + 8 * kh) * S_LD + key] = v;
        rmax[r] = fmaxf(rmax[r], v);
      }
    }
    // half-group (16-lane) max reduce; row m=r+8*kh lives in lanes kh*16..kh*16+15
#pragma unroll
    for (int off = 1; off < 16; off <<= 1)
#pragma unroll
      for (int r = 0; r < 8; ++r) rmax[r] = fmaxf(rmax[r], __shfl_xor(rmax[r], off, 32));

    // ---- pass 2: exponentiate in LDS, accumulate row sums ----
    float sm[8] = {0.f, 0.f, 0.f, 0.f, 0.f, 0.f, 0.f, 0.f};
#pragma unroll 2
    for (int nc = 0; nc < 20; ++nc) {
#pragma unroll
      for (int r = 0; r < 8; ++r) {
        int sidx = (r + 8 * kh) * S_LD + nc * 16 + n16;
        float e = __expf(S[sidx] - rmax[r]);
        S[sidx] = e;
        sm[r] += e;
      }
    }
#pragma unroll
    for (int off = 1; off < 16; off <<= 1)
#pragma unroll
      for (int r = 0; r < 8; ++r) sm[r] += __shfl_xor(sm[r], off, 32);
    float inv[8];
#pragma unroll
    for (int r = 0; r < 8; ++r) inv[r] = 1.0f / sm[r];
    // publish per-row 1/sum for A-layout consumers (same wave, LDS ops stay in order)
    if (n16 == 0) {
#pragma unroll
      for (int r = 0; r < 8; ++r) rowinv[r + 8 * kh] = inv[r];
    }

    // ---- pass 3: write normalized attn to corrmap (required output) ----
    float* cm = corrmap + ((size_t)(b * NHEADS + h) * NTOK + row0) * NTOK;
#pragma unroll 2
    for (int nc = 0; nc < 20; ++nc)
#pragma unroll
      for (int r = 0; r < 8; ++r)
        cm[(size_t)(r + 8 * kh) * NTOK + nc * 16 + n16] =
            S[(r + 8 * kh) * S_LD + nc * 16 + n16] * inv[r];

    // ---- attn @ V : A from LDS scores (row-scaled during f16 convert), B from LDS vT ----
    float invm = rowinv[n16];   // A-layout row m = lane&15 (both half-groups)
#pragma unroll
    for (int dc = 0; dc < 4; ++dc) {
      v8f o = {};
#pragma unroll 2
      for (int tc = 0; tc < 10; ++tc) {
        const float* pa = S + n16 * S_LD + tc * 32 + kh * 8;   // 16B aligned
        v4f p0 = *(const v4f*)pa;
        v4f p1 = *(const v4f*)(pa + 4);
        v4f p2 = *(const v4f*)(pa + 16);
        v4f p3 = *(const v4f*)(pa + 20);
        v16h a;
#pragma unroll
        for (int i = 0; i < 4; ++i) {
          a[i]      = (_Float16)(p0[i] * invm);
          a[i + 4]  = (_Float16)(p1[i] * invm);
          a[i + 8]  = (_Float16)(p2[i] * invm);
          a[i + 12] = (_Float16)(p3[i] * invm);
        }
        v16h bv = ldB(vT + (dc * 16 + n16) * VT_LD, VT_LD, tc * 32, 0, kh);
        o = WMMA_F16(a, bv, o);
      }
      _Float16* po = ao16 + (rowBase + row0) * DIM + h * HEAD_D + dc * 16;
#pragma unroll
      for (int r = 0; r < 8; ++r)
        po[(size_t)(r + 8 * kh) * DIM + n16] = (_Float16)o[r];
    }
  }
}

// ---------------- top-k (exact argsort of 256 scores, descending, stable) ----------------
__global__ __launch_bounds__(256) void topk_kernel(
    const float* __restrict__ cm_rgb, const float* __restrict__ cm_tir,
    const unsigned char* __restrict__ tmpl, const int* __restrict__ gidx_s,
    int* __restrict__ topk, float* __restrict__ out_gisn, float* __restrict__ out_rem) {
  int b = blockIdx.x;
  int s = threadIdx.x;
  __shared__ float key[256];
  __shared__ int   idx[256];
  __shared__ float denom_sh;
  if (s == 0) {
    float d = 0.f;
    for (int t = 0; t < LENS_T; ++t) d += tmpl[b * LENS_T + t] ? 1.0f : 0.0f;
    denom_sh = d;
  }
  float acc = 0.f;
  const size_t hb = (size_t)b * NHEADS;
  for (int h = 0; h < NHEADS; ++h) {
    const float* pr = cm_rgb + ((hb + h) * NTOK) * NTOK + LENS_T + s;
    const float* pt = cm_tir + ((hb + h) * NTOK) * NTOK + LENS_T + s;
    float hs = 0.f;
    for (int t = 0; t < LENS_T; ++t) {
      float w = tmpl[b * LENS_T + t] ? 1.0f : 0.0f;
      hs += (pr[(size_t)t * NTOK] + pt[(size_t)t * NTOK]) * w;
    }
    acc += hs;
  }
  __syncthreads();
  acc = acc / denom_sh * (1.0f / NHEADS);
  key[s] = acc;
  idx[s] = s;
  __syncthreads();

  // bitonic sort: "before" = higher score, tie -> lower index (stable argsort of -scores)
  for (int k = 2; k <= 256; k <<= 1) {
    for (int j = k >> 1; j > 0; j >>= 1) {
      int ixj = s ^ j;
      if (ixj > s) {
        bool up = ((s & k) == 0);
        float k1 = key[s], k2 = key[ixj];
        int   i1 = idx[s], i2 = idx[ixj];
        bool before = (k1 > k2) || (k1 == k2 && i1 < i2);
        if (up ? !before : before) {
          key[s] = k2; key[ixj] = k1;
          idx[s] = i2; idx[ixj] = i1;
        }
      }
      __syncthreads();
    }
  }
  topk[b * 256 + s] = idx[s];
  if (s < NKEEP) out_gisn[b * NKEEP + s] = (float)gidx_s[b * LENS_S + idx[s]];
  else           out_rem[b * NREM + (s - NKEEP)] = (float)gidx_s[b * LENS_S + idx[s]];
}

// ---------------- token gather ----------------
__global__ void gather_kernel(const float* __restrict__ xa, const int* __restrict__ topk,
                              float* __restrict__ out) {
  int i = blockIdx.x * 256 + threadIdx.x;
  if (i >= MNEW * DIM) return;
  int c = i % DIM;
  int r = (i / DIM) % NNEW;
  int b = i / (DIM * NNEW);
  int src = (r < LENS_T) ? r : (LENS_T + topk[b * 256 + (r - LENS_T)]);
  out[i] = xa[((size_t)b * NTOK + src) * DIM + c];
}

// ---------------- misc copies (global_index_t, js_loss) ----------------
__global__ void misc_copy_kernel(const int* __restrict__ git, const float* __restrict__ js,
                                 float* __restrict__ out_git, float* __restrict__ out_js) {
  int i = blockIdx.x * 256 + threadIdx.x;
  if (i < BATCH * LENS_T) out_git[i] = (float)git[i];
  if (i == 0) out_js[0] = js[0];
}

// ==================== host launcher ====================
extern "C" void kernel_launch(void* const* d_in, const int* in_sizes, int n_in,
                              void* d_out, int out_size, void* d_ws, size_t ws_size,
                              hipStream_t stream) {
  (void)in_sizes; (void)n_in; (void)out_size; (void)ws_size;
  const float* x_rgb  = (const float*)d_in[0];
  const float* x_tir  = (const float*)d_in[1];
  const int*   git    = (const int*)d_in[2];
  const int*   gis    = (const int*)d_in[3];
  const unsigned char* mask = (const unsigned char*)d_in[4];
  const unsigned char* tmpl = (const unsigned char*)d_in[5];
  const float* js     = (const float*)d_in[7];
  const float* g1     = (const float*)d_in[8];
  const float* b1     = (const float*)d_in[9];
  const float* w_qkv  = (const float*)d_in[10];
  const float* w_proj = (const float*)d_in[11];
  const float* b_proj = (const float*)d_in[12];
  const float* g2     = (const float*)d_in[13];
  const float* b2     = (const float*)d_in[14];
  const float* w_fc1  = (const float*)d_in[15];
  const float* b_fc1  = (const float*)d_in[16];
  const float* w_fc2  = (const float*)d_in[17];
  const float* b_fc2  = (const float*)d_in[18];

  // ---- d_out layout (floats, tuple concatenated) ----
  float* out = (float*)d_out;
  float* o_xrgb = out;                                   // 32*244*768
  float* o_xtir = o_xrgb + (size_t)MNEW * DIM;           // 32*244*768
  float* o_git  = o_xtir + (size_t)MNEW * DIM;           // 2048
  float* o_gisn = o_git + BATCH * LENS_T;                // 5760
  float* o_rem  = o_gisn + BATCH * NKEEP;                // 2432
  float* o_cmr  = o_rem + BATCH * NREM;                  // 39,321,600
  float* o_cmt  = o_cmr + (size_t)BATCH * NHEADS * NTOK * NTOK;
  float* o_js   = o_cmt + (size_t)BATCH * NHEADS * NTOK * NTOK;

  // ---- workspace layout (bytes, 256-aligned; BIG region aliased QKV16 -> H16) ----
  char* w = (char*)d_ws;
  _Float16* WQ16  = (_Float16*)(w + 0);                  //  3,538,944 B
  _Float16* WP16  = (_Float16*)(w + 3538944);            //  1,179,648 B
  _Float16* WF1   = (_Float16*)(w + 4718592);            //  4,718,592 B
  _Float16* WF2   = (_Float16*)(w + 9437184);            //  4,718,592 B
  _Float16* XN16  = (_Float16*)(w + 14155776);           // 15,728,640 B (aliased by XN2 later)
  _Float16* BIG   = (_Float16*)(w + 29884416);           // 47,972,352 B (QKV16, then H16)
  _Float16* AO16  = (_Float16*)(w + 77856768);           // 15,728,640 B
  float*    XA0   = (float*)   (w + 93585408);           // 31,457,280 B
  float*    XA1   = (float*)   (w + 125042688);          // 31,457,280 B
  int*      TOPK  = (int*)     (w + 156499968);          //     32,768 B
  _Float16* QKV16 = BIG;
  _Float16* H16   = BIG;
  _Float16* XN2   = XN16;

  // ---- weights -> f16 ----
  f32_to_f16_kernel<<<(3 * DIM * DIM + 255) / 256, 256, 0, stream>>>(w_qkv, WQ16, 3 * DIM * DIM);
  f32_to_f16_kernel<<<(DIM * DIM + 255) / 256, 256, 0, stream>>>(w_proj, WP16, DIM * DIM);
  f32_to_f16_kernel<<<(MLP_H * DIM + 255) / 256, 256, 0, stream>>>(w_fc1, WF1, MLP_H * DIM);
  f32_to_f16_kernel<<<(DIM * MLP_H + 255) / 256, 256, 0, stream>>>(w_fc2, WF2, DIM * MLP_H);

  // ---- attention phase, per stream (serial reuse of XN16/QKV16/AO16) ----
  const float* xs[2]  = {x_rgb, x_tir};
  float*       cms[2] = {o_cmr, o_cmt};
  float*       xas[2] = {XA0, XA1};
  for (int s = 0; s < 2; ++s) {
    layernorm_f16_kernel<<<MTOK, 256, 0, stream>>>(xs[s], g1, b1, XN16);
    gemm_wmma_kernel<true, false><<<dim3(MTOK / 64, (3 * DIM) / 256), 256, 0, stream>>>(
        XN16, WQ16, nullptr, nullptr, QKV16, nullptr, MTOK, 3 * DIM, DIM);
    attention_kernel<<<BATCH * NHEADS, 256, ATT_SMEM, stream>>>(QKV16, mask, cms[s], AO16);
    gemm_wmma_kernel<false, false><<<dim3(MTOK / 64, DIM / 256), 256, 0, stream>>>(
        AO16, WP16, b_proj, xs[s], nullptr, xas[s], MTOK, DIM, DIM);
  }

  // ---- candidate top-k + index outputs ----
  topk_kernel<<<BATCH, 256, 0, stream>>>(o_cmr, o_cmt, tmpl, gis, TOPK, o_gisn, o_rem);

  // ---- gather + MLP phase, per stream (BIG now reused as H16) ----
  float* oxs[2] = {o_xrgb, o_xtir};
  for (int s = 0; s < 2; ++s) {
    gather_kernel<<<(MNEW * DIM + 255) / 256, 256, 0, stream>>>(xas[s], TOPK, oxs[s]);
    layernorm_f16_kernel<<<MNEW, 256, 0, stream>>>(oxs[s], g2, b2, XN2);
    gemm_wmma_kernel<true, true><<<dim3(MNEW / 64, MLP_H / 256), 256, 0, stream>>>(
        XN2, WF1, b_fc1, nullptr, H16, nullptr, MNEW, MLP_H, DIM);
    gemm_wmma_kernel<false, false><<<dim3(MNEW / 64, DIM / 256), 256, 0, stream>>>(
        H16, WF2, b_fc2, oxs[s], nullptr, oxs[s], MNEW, DIM, MLP_H);
  }

  // ---- passthrough outputs ----
  misc_copy_kernel<<<(BATCH * LENS_T + 255) / 256, 256, 0, stream>>>(git, js, o_git, o_js);
}
